// Mamba3DBlock_68436008895053
// MI455X (gfx1250) — compile-verified, matrix operand order checked by reading, not run on hardware
//
#include <hip/hip_runtime.h>
#include <cstdint>
#include <cstddef>

// ---- problem constants (from reference) ----
#define DM    1024      // D_MODEL
#define DS    16        // D_STATE
#define DCONV 4
#define DI    2048      // D_INNER
#define DTR   64        // DT_RANK
#define NX    (DTR + 2*DS)   // 96
#define BB    4
#define LL    2048
#define MROWS (BB*LL)        // 8192

typedef __attribute__((ext_vector_type(16))) __bf16 v16bf;
typedef __attribute__((ext_vector_type(8)))  __bf16 v8bf;
typedef __attribute__((ext_vector_type(8)))  float  v8f;

__device__ __forceinline__ __bf16 f2bf(float f) {
  uint32_t u = __builtin_bit_cast(uint32_t, f);
  uint32_t r = u + 0x7FFFu + ((u >> 16) & 1u);   // round-to-nearest-even
  uint16_t h = (uint16_t)(r >> 16);
  return __builtin_bit_cast(__bf16, h);
}
__device__ __forceinline__ float sigmoidf_(float x) { return 1.f / (1.f + __expf(-x)); }

// Load one 16-wide bf16 WMMA fragment slice: two contiguous 16B groups
// (K-groups {ka..ka+7} and {ka+16..ka+23}), per the ISA 16-bit A/B layout.
__device__ __forceinline__ v16bf ldfrag(const __bf16* p) {
  v8bf lo = *(const v8bf*)(p);
  v8bf hi = *(const v8bf*)(p + 16);
  return __builtin_shufflevector(lo, hi, 0,1,2,3,4,5,6,7,8,9,10,11,12,13,14,15);
}

// ---------------- f32 [K,N] -> bf16 transposed [N,K] (one-time weight prep) ----------------
__global__ void f32_to_bf16_T_kernel(const float* __restrict__ s, __bf16* __restrict__ d,
                                     int K, int N) {
  size_t i = (size_t)blockIdx.x * blockDim.x + threadIdx.x;
  if (i >= (size_t)K * N) return;
  int k = (int)(i / N);
  int n = (int)(i % N);
  d[(size_t)n * K + k] = f2bf(s[i]);
}

// ---------------- LayerNorm -> bf16 ----------------
__global__ void ln_bf16_kernel(const float* __restrict__ x, const float* __restrict__ w,
                               const float* __restrict__ b, __bf16* __restrict__ xn) {
  __shared__ float s1[256], s2[256];
  int rowi = blockIdx.x;
  const float* xr = x + (size_t)rowi * DM;
  float a = 0.f, q = 0.f;
  for (int i = threadIdx.x; i < DM; i += 256) { float v = xr[i]; a += v; q += v * v; }
  s1[threadIdx.x] = a; s2[threadIdx.x] = q;
  __syncthreads();
  for (int st = 128; st > 0; st >>= 1) {
    if ((int)threadIdx.x < st) { s1[threadIdx.x] += s1[threadIdx.x + st]; s2[threadIdx.x] += s2[threadIdx.x + st]; }
    __syncthreads();
  }
  float mu  = s1[0] / (float)DM;
  float var = s2[0] / (float)DM - mu * mu;
  float rs  = rsqrtf(var + 1e-5f);
  for (int i = threadIdx.x; i < DM; i += 256)
    xn[(size_t)rowi * DM + i] = f2bf((xr[i] - mu) * rs * w[i] + b[i]);
}

// ---- bf16 WMMA GEMM: (16*TM)x(16*TN) block per wave, TM*TN WMMAs per K-step ----
// A:  [M,K]  bf16 row-major (lda = row stride)
// Bt: [N,K]  bf16 row-major (transposed weights -> contiguous fragment loads)
// MODE 0: C0 = acc (f32); optional bf16 copy in Cbf
// MODE 1: split columns at Nsplit -> C0 / C1 (both row stride Nsplit)
// MODE 2: C0 = softplus(acc + bias[col])
// MODE 3: C0 = resid[row,col] + acc
template <int MODE, int TM, int TN>
__global__ __launch_bounds__(128, 1)
void gemm_bf16_wmma(const __bf16* __restrict__ A, int lda,
                    const __bf16* __restrict__ Bt,
                    int M, int N, int K,
                    float* __restrict__ C0, float* __restrict__ C1, int Nsplit,
                    __bf16* __restrict__ Cbf,
                    const float* __restrict__ bias,
                    const float* __restrict__ resid) {
  int wave  = blockIdx.x * (blockDim.x >> 5) + ((int)threadIdx.x >> 5);
  int nBlkN = N / (16 * TN);
  int m0 = (wave / nBlkN) * (16 * TM);
  int n0 = (wave % nBlkN) * (16 * TN);
  if (m0 >= M) return;

  int lane = threadIdx.x & 31;
  int half = lane >> 4;
  int r    = lane & 15;

  const __bf16* arow[TM];
  const __bf16* brow[TN];
#pragma unroll
  for (int i = 0; i < TM; ++i) arow[i] = A  + (size_t)(m0 + i * 16 + r) * lda;
#pragma unroll
  for (int j = 0; j < TN; ++j) brow[j] = Bt + (size_t)(n0 + j * 16 + r) * K;

  v8f acc[TM][TN];
#pragma unroll
  for (int i = 0; i < TM; ++i)
#pragma unroll
    for (int j = 0; j < TN; ++j) acc[i][j] = (v8f){};

  for (int k0 = 0; k0 < K; k0 += 32) {
    int ka = k0 + half * 8;
    v16bf af[TM], bf[TN];
#pragma unroll
    for (int i = 0; i < TM; ++i) af[i] = ldfrag(arow[i] + ka);
#pragma unroll
    for (int j = 0; j < TN; ++j) bf[j] = ldfrag(brow[j] + ka);
    // prefetch next K-chunk of the streaming A panel into near caches
#pragma unroll
    for (int i = 0; i < TM; ++i) __builtin_prefetch(arow[i] + ka + 32, 0, 3);
#pragma unroll
    for (int i = 0; i < TM; ++i)
#pragma unroll
      for (int j = 0; j < TN; ++j)
        acc[i][j] = __builtin_amdgcn_wmma_f32_16x16x32_bf16(
            false, af[i], false, bf[j], (short)0, acc[i][j], false, false);
  }

#pragma unroll
  for (int ti = 0; ti < TM; ++ti) {
#pragma unroll
    for (int tj = 0; tj < TN; ++tj) {
      int orow = m0 + ti * 16 + half * 8;   // C/D layout: M = 8*half + j
      int ocol = n0 + tj * 16 + r;          //             N = lane & 15
#pragma unroll
      for (int j = 0; j < 8; ++j) {
        float v = acc[ti][tj][j];
        size_t rr = (size_t)(orow + j);
        if (MODE == 0) {
          C0[rr * N + ocol] = v;
          if (Cbf) Cbf[rr * N + ocol] = f2bf(v);
        } else if (MODE == 1) {
          if (ocol < Nsplit) C0[rr * Nsplit + ocol] = v;
          else               C1[rr * Nsplit + (ocol - Nsplit)] = v;
        } else if (MODE == 2) {
          float s = v + bias[ocol];
          C0[rr * N + ocol] = (s > 20.f) ? s : log1pf(__expf(s));
        } else { // MODE 3
          C0[rr * N + ocol] = resid[rr * N + ocol] + v;
        }
      }
    }
  }
}

// ---------------- causal depthwise conv (k=4) + bias + SiLU ----------------
__global__ void conv_silu_kernel(const float* __restrict__ xin, const float* __restrict__ cw,
                                 const float* __restrict__ cb, float* __restrict__ xc,
                                 __bf16* __restrict__ xcb) {
  size_t idx = (size_t)blockIdx.x * blockDim.x + threadIdx.x;
  if (idx >= (size_t)MROWS * DI) return;
  int d = (int)(idx % DI);
  size_t bl = idx / DI;           // b*LL + l
  int l = (int)(bl % LL);
  float acc = cb[d];
#pragma unroll
  for (int j = 0; j < DCONV; ++j) {
    int ll = l - (DCONV - 1) + j;
    if (ll >= 0) acc += xin[(bl + (size_t)(ll - l)) * DI + d] * cw[d * DCONV + j];
  }
  float v = acc * sigmoidf_(acc);
  xc[idx]  = v;
  xcb[idx] = f2bf(v);
}

// ---------------- selective scan: 16 lanes (one per state) per (b,d) ----------------
__global__ void scan_kernel(const float* __restrict__ dtp, const float* __restrict__ xc,
                            const float* __restrict__ xdbl, const float* __restrict__ A_log,
                            float* __restrict__ ys) {
  int gt = blockIdx.x * blockDim.x + threadIdx.x;
  int g = gt >> 4;           // which (b,d)
  int n = gt & 15;           // state index
  if (g >= BB * DI) return;
  int b = g / DI, d = g % DI;
  float Av = -__expf(A_log[d * DS + n]);
  float h = 0.f;
  for (int l = 0; l < LL; ++l) {
    size_t bl = (size_t)b * LL + l;
    float dtv = dtp[bl * DI + d];
    float u   = xc[bl * DI + d];
    const float* xr = xdbl + bl * NX;
    float Bn = xr[DTR + n];
    float Cn = xr[DTR + DS + n];
    h = h * __expf(dtv * Av) + dtv * Bn * u;
    float yv = h * Cn;
    yv += __shfl_xor(yv, 1, 32);
    yv += __shfl_xor(yv, 2, 32);
    yv += __shfl_xor(yv, 4, 32);
    yv += __shfl_xor(yv, 8, 32);
    if (n == 0) ys[bl * DI + d] = yv;
  }
}

// ---------------- gate: y = (ys + xc*D) * silu(z) -> bf16 ----------------
__global__ void gate_kernel(const float* __restrict__ ys, const float* __restrict__ xc,
                            const float* __restrict__ z, const float* __restrict__ Dp,
                            __bf16* __restrict__ yb) {
  size_t idx = (size_t)blockIdx.x * blockDim.x + threadIdx.x;
  if (idx >= (size_t)MROWS * DI) return;
  int d = (int)(idx % DI);
  float zv = z[idx];
  float y = (ys[idx] + xc[idx] * Dp[d]) * (zv * sigmoidf_(zv));
  yb[idx] = f2bf(y);
}

// ---------------- host launcher ----------------
extern "C" void kernel_launch(void* const* d_in, const int* in_sizes, int n_in,
                              void* d_out, int out_size, void* d_ws, size_t ws_size,
                              hipStream_t stream) {
  const float* x      = (const float*)d_in[0];
  const float* ln_w   = (const float*)d_in[1];
  const float* ln_b   = (const float*)d_in[2];
  const float* W_in   = (const float*)d_in[3];
  const float* conv_w = (const float*)d_in[4];
  const float* conv_b = (const float*)d_in[5];
  const float* W_xp   = (const float*)d_in[6];
  const float* W_dt   = (const float*)d_in[7];
  const float* b_dt   = (const float*)d_in[8];
  const float* A_log  = (const float*)d_in[9];
  const float* D_par  = (const float*)d_in[10];
  const float* W_out  = (const float*)d_in[11];
  (void)in_sizes; (void)n_in; (void)out_size; (void)ws_size;

  char* ws = (char*)d_ws;
  size_t off = 0;
  auto take = [&](size_t bytes) -> void* {
    void* p = ws + off;
    off += (bytes + 255) & ~(size_t)255;
    return p;
  };

  __bf16* xn_bf   = (__bf16*)take((size_t)MROWS * DM * 2);
  __bf16* WinT    = (__bf16*)take((size_t)DM * 2 * DI * 2);   // [2*DI, DM]
  __bf16* WxpT    = (__bf16*)take((size_t)DI * NX * 2);       // [NX, DI]
  __bf16* WdtT    = (__bf16*)take((size_t)DTR * DI * 2);      // [DI, DTR]
  __bf16* WoutT   = (__bf16*)take((size_t)DI * DM * 2);       // [DM, DI]
  float*  xin     = (float*) take((size_t)MROWS * DI * 4);
  float*  zf      = (float*) take((size_t)MROWS * DI * 4);
  float*  xcf     = (float*) take((size_t)MROWS * DI * 4);
  __bf16* xcb     = (__bf16*)take((size_t)MROWS * DI * 2);
  float*  xdbl    = (float*) take((size_t)MROWS * NX * 4);
  __bf16* xdblb   = (__bf16*)take((size_t)MROWS * NX * 2);
  float*  dtf     = (float*) take((size_t)MROWS * DI * 4);
  float*  ysf     = (float*) take((size_t)MROWS * DI * 4);
  __bf16* ybf     = (__bf16*)take((size_t)MROWS * DI * 2);

  // 1) weights -> bf16, transposed to [N,K] for contiguous WMMA B-fragment loads
  auto cvtT = [&](const float* s, __bf16* d, int K, int N) {
    size_t n = (size_t)K * N;
    f32_to_bf16_T_kernel<<<(unsigned)((n + 255) / 256), 256, 0, stream>>>(s, d, K, N);
  };
  cvtT(W_in,  WinT,  DM,  2 * DI);
  cvtT(W_xp,  WxpT,  DI,  NX);
  cvtT(W_dt,  WdtT,  DTR, DI);
  cvtT(W_out, WoutT, DI,  DM);

  // 2) LayerNorm -> bf16
  ln_bf16_kernel<<<MROWS, 256, 0, stream>>>(x, ln_w, ln_b, xn_bf);

  // 3) xz = xn @ W_in  (split into x_in | z), 64x32 per wave
  {
    int waves = (MROWS / 64) * ((2 * DI) / 32);
    gemm_bf16_wmma<1, 4, 2><<<waves / 4, 128, 0, stream>>>(xn_bf, DM, WinT, MROWS, 2 * DI, DM,
                                                           xin, zf, DI, nullptr, nullptr, nullptr);
  }

  // 4) causal depthwise conv + bias + SiLU
  size_t nel = (size_t)MROWS * DI;
  conv_silu_kernel<<<(unsigned)((nel + 255) / 256), 256, 0, stream>>>(xin, conv_w, conv_b, xcf, xcb);

  // 5) xdbl = xc @ W_xproj  (skinny N=96 -> 32x32 per wave)
  {
    int waves = (MROWS / 32) * (NX / 32);
    gemm_bf16_wmma<0, 2, 2><<<waves / 4, 128, 0, stream>>>(xcb, DI, WxpT, MROWS, NX, DI,
                                                           xdbl, nullptr, 0, xdblb, nullptr, nullptr);
  }

  // 6) dt = softplus(dt_r @ W_dt + b_dt)   (A = xdbl cols 0..63, lda = 96), 64x32 per wave
  {
    int waves = (MROWS / 64) * (DI / 32);
    gemm_bf16_wmma<2, 4, 2><<<waves / 4, 128, 0, stream>>>(xdblb, NX, WdtT, MROWS, DI, DTR,
                                                           dtf, nullptr, 0, nullptr, b_dt, nullptr);
  }

  // 7) selective scan
  scan_kernel<<<(BB * DI * 16) / 256, 256, 0, stream>>>(dtf, xcf, xdbl, A_log, ysf);

  // 8) gate
  gate_kernel<<<(unsigned)((nel + 255) / 256), 256, 0, stream>>>(ysf, xcf, zf, D_par, ybf);

  // 9) out = x + y @ W_out, 64x32 per wave
  {
    int waves = (MROWS / 64) * (DM / 32);
    gemm_bf16_wmma<3, 4, 2><<<waves / 4, 128, 0, stream>>>(ybf, DI, WoutT, MROWS, DM, DI,
                                                           (float*)d_out, nullptr, 0, nullptr, nullptr, x);
  }
}
